// AttentionFusion_77025943486638
// MI455X (gfx1250) — matrix operand order repaired and checked by reading, not compiled, not run on hardware
//
#include <hip/hip_runtime.h>
#include <math.h>

// ---------------------------------------------------------------------------
// MI455X (gfx1250). bf16 WMMA GEMMs with double-buffered LDS fed by
// GLOBAL_LOAD_ASYNC_TO_LDS_B128 (ASYNCcnt). KB=64: 16 WMMAs per barrier pair.
// All GEMM operands are pre-converted/produced as bf16 in global memory; f32
// kept only for residuals, layernorm inputs and the final combine. Cross-attn
// with seq_len==1 is algebraically reduced (softmax over one key == 1 ->
// V/O projections only). Workspace ~730 MB, bump-allocated from d_ws.
// ---------------------------------------------------------------------------

#define BATCH 8192
#define H_DIM 1024

typedef __attribute__((ext_vector_type(16))) __bf16 v16bf;
typedef __attribute__((ext_vector_type(8)))  __bf16 v8bf;
typedef __attribute__((ext_vector_type(8)))  float  v8f;

#define BM 128
#define BN 128
#define KB 64
#define LDS_STRIDE (KB + 8)             // 72 bf16 = 144 B/row (16B-aligned)
#define BUFBYTES (BM * LDS_STRIDE * 2)  // one LDS tile buffer (18432 B)

__device__ __forceinline__ __bf16 f2bf(float f) {
  unsigned u = __builtin_bit_cast(unsigned, f);
  unsigned r = u + 0x7FFFu + ((u >> 16) & 1u);   // round-to-nearest-even
  unsigned short h = (unsigned short)(r >> 16);
  return __builtin_bit_cast(__bf16, h);
}
__device__ __forceinline__ float bf2f(__bf16 h) {
  unsigned short s = __builtin_bit_cast(unsigned short, h);
  unsigned u = (unsigned)s << 16;
  return __builtin_bit_cast(float, u);
}

// Async DMA: global -> LDS, 16 bytes per lane, tracked by ASYNCcnt.
__device__ __forceinline__ void async_b128(unsigned lds_addr, unsigned voff,
                                           const void* sbase) {
  asm volatile("global_load_async_to_lds_b128 %0, %1, %2"
               :: "v"(lds_addr), "v"(voff), "s"(sbase) : "memory");
}
__device__ __forceinline__ void wait_async0() {
  asm volatile("s_wait_asynccnt 0" ::: "memory");
}

// A fragment (16x32 bf16, MxK): lane L holds row M=L%16, lane-group g=L/16:
// halves[0..7]=K=8g..8g+7, halves[8..15]=K=16+8g..16+8g+7 (ISA 7.12.2).
__device__ __forceinline__ v16bf load_fragA(const __bf16* tile, int lane) {
  const int r = lane & 15;
  const int g = lane >> 4;
  const __bf16* p = tile + r * LDS_STRIDE + g * 8;
  v8bf lo = *(const v8bf*)(p);
  v8bf hi = *(const v8bf*)(p + 16);
  return __builtin_shufflevector(lo, hi, 0,1,2,3,4,5,6,7,8,9,10,11,12,13,14,15);
}
// B fragment (32x16 bf16, KxN): lane L holds column N=L%16, lane-group g holds
// K=8g..8g+7 and 16+8g..16+8g+7 (same as A); W staged row-major (NxK).
__device__ __forceinline__ v16bf load_fragB(const __bf16* tile, int lane) {
  const int r = lane & 15;
  const int g = lane >> 4;
  const __bf16* p = tile + r * LDS_STRIDE + g * 8;
  v8bf lo = *(const v8bf*)(p);
  v8bf hi = *(const v8bf*)(p + 16);
  return __builtin_shufflevector(lo, hi, 0,1,2,3,4,5,6,7,8,9,10,11,12,13,14,15);
}

// C = act(A @ W^T + bias [+ Res]).  A: MxK bf16, W: NxK bf16 row-major.
// Split-A: k < kSplit -> A, else A2 (requires lda2 == lda when split used;
// kSplit is a multiple of KB so each staged tile is wholly on one side).
// M == 8192, N % 128 == 0, K % 64 == 0 at every call site.
// ACT: 0 none, 1 exact GELU, 2 sigmoid.  WF32/WBF: write f32 / bf16 output.
template <int ACT, bool HAS_RES, bool WF32, bool WBF>
__global__ __launch_bounds__(256) void gemm_bias_act(
    const __bf16* __restrict__ A,  int lda,
    const __bf16* __restrict__ A2, int kSplit,
    const __bf16* __restrict__ W,
    const float*  __restrict__ bias,
    const float*  __restrict__ Res,
    float*        __restrict__ C,
    __bf16*       __restrict__ Cbf,
    int N, int K)
{
  __shared__ __align__(16) __bf16 As[2][BM * LDS_STRIDE];
  __shared__ __align__(16) __bf16 Bs[2][BN * LDS_STRIDE];

  const int tid    = threadIdx.x;
  const int lane   = tid & 31;
  const int wave   = tid >> 5;          // 8 waves
  const int waveM  = (wave & 3) * 32;   // 4x2 wave grid, 32x64 per wave
  const int waveN  = (wave >> 2) * 64;
  const int blockM = blockIdx.y * BM;
  const int blockN = blockIdx.x * BN;

  // Staging geometry: tile = 128 rows x 128 B; chunk c (0..1023) = 16 B.
  // Each wave issues 4 async b128 per tile (A and B): chunks wave*128+i*32+lane.
  unsigned aoff[4], boff[4], ldsb[4];
  #pragma unroll
  for (int i = 0; i < 4; ++i) {
    const int c = wave * 128 + i * 32 + lane;
    const int r = c >> 3;
    const int s = (c & 7) * 8;          // element offset within row
    aoff[i] = (unsigned)(r * lda + s) * 2u;
    boff[i] = (unsigned)(r * K + s) * 2u;
    ldsb[i] = (unsigned)(r * LDS_STRIDE + s) * 2u;
  }
  const unsigned asBase = (unsigned)(unsigned long long)&As[0][0];
  const unsigned bsBase = (unsigned)(unsigned long long)&Bs[0][0];

  const __bf16* Abase  = A  + (size_t)blockM * lda;
  const __bf16* A2base = A2 + (size_t)blockM * lda;   // lda2 == lda by contract
  const __bf16* Bbase  = W  + (size_t)blockN * K;

  auto issue = [&](int buf, int k0) {
    const __bf16* bA  = (k0 < kSplit) ? Abase : A2base;
    const unsigned ko2 = (unsigned)(((k0 < kSplit) ? k0 : (k0 - kSplit)) * 2);
    const unsigned kb2 = (unsigned)(k0 * 2);
    const unsigned aB = asBase + (unsigned)buf * BUFBYTES;
    const unsigned bB = bsBase + (unsigned)buf * BUFBYTES;
    #pragma unroll
    for (int i = 0; i < 4; ++i) async_b128(aB + ldsb[i], aoff[i] + ko2, bA);
    #pragma unroll
    for (int i = 0; i < 4; ++i) async_b128(bB + ldsb[i], boff[i] + kb2, Bbase);
  };

  const v8f vzero = {0.f,0.f,0.f,0.f,0.f,0.f,0.f,0.f};
  v8f acc[2][4];
  #pragma unroll
  for (int mi = 0; mi < 2; ++mi)
    #pragma unroll
    for (int ni = 0; ni < 4; ++ni) acc[mi][ni] = vzero;

  int buf = 0;
  issue(0, 0);
  for (int k0 = 0; k0 < K; k0 += KB, buf ^= 1) {
    wait_async0();        // this wave's writes into `buf` are in LDS
    __syncthreads();      // everyone's are
    if (k0 + KB < K) issue(buf ^ 1, k0 + KB);   // overlap DMA with compute

    #pragma unroll
    for (int ks = 0; ks < KB; ks += 32) {       // two 32-K sub-steps
      v16bf af[2], bfr[4];
      #pragma unroll
      for (int mi = 0; mi < 2; ++mi)
        af[mi] = load_fragA(&As[buf][(waveM + mi * 16) * LDS_STRIDE + ks], lane);
      #pragma unroll
      for (int ni = 0; ni < 4; ++ni)
        bfr[ni] = load_fragB(&Bs[buf][(waveN + ni * 16) * LDS_STRIDE + ks], lane);

      #pragma unroll
      for (int mi = 0; mi < 2; ++mi)
        #pragma unroll
        for (int ni = 0; ni < 4; ++ni)
          acc[mi][ni] = __builtin_amdgcn_wmma_f32_16x16x32_bf16(
              false, af[mi], false, bfr[ni], (short)0, acc[mi][ni], false, false);
    }

    __syncthreads();      // all reads of `buf` done before it is re-filled
  }

  // Epilogue. D layout: lane L, vgpr v -> M = v + 8*(L/16), N = L%16.
  const int lr = lane & 15;
  const int lg = lane >> 4;
  #pragma unroll
  for (int mi = 0; mi < 2; ++mi) {
    #pragma unroll
    for (int ni = 0; ni < 4; ++ni) {
      const int n  = blockN + waveN + ni * 16 + lr;
      const int mB = blockM + waveM + mi * 16 + lg * 8;
      const float bv = bias[n];
      float r[8];
      if (HAS_RES) {
        const float* rp = Res + (size_t)mB * N + n;
        #pragma unroll
        for (int v = 0; v < 8; ++v) r[v] = rp[(size_t)v * N];  // clause of 8
      }
      float*  cp = C   + (size_t)mB * N + n;
      __bf16* bp = Cbf + (size_t)mB * N + n;
      #pragma unroll
      for (int v = 0; v < 8; ++v) {
        float x = acc[mi][ni][v] + bv;
        if (HAS_RES) x += r[v];
        if (ACT == 1)      x = 0.5f * x * (1.0f + erff(x * 0.70710678118654752f));
        else if (ACT == 2) x = 1.0f / (1.0f + expf(-x));
        if (WF32) cp[(size_t)v * N] = x;
        if (WBF)  bp[(size_t)v * N] = f2bf(x);
      }
    }
  }
}

// f32 -> bf16 bulk convert (n % 8 == 0).
__global__ __launch_bounds__(256) void cvt_bf16(
    const float* __restrict__ x, __bf16* __restrict__ y, long n)
{
  long i = ((long)blockIdx.x * 256 + threadIdx.x) * 8;
  if (i >= n) return;
  float4 a = *(const float4*)(x + i);
  float4 b = *(const float4*)(x + i + 4);
  v8bf o;
  o[0]=f2bf(a.x); o[1]=f2bf(a.y); o[2]=f2bf(a.z); o[3]=f2bf(a.w);
  o[4]=f2bf(b.x); o[5]=f2bf(b.y); o[6]=f2bf(b.z); o[7]=f2bf(b.w);
  *(v8bf*)(y + i) = o;
}

// Row layernorm over D=1024; writes f32 (for residual use) + bf16 (for GEMMs).
__global__ __launch_bounds__(256) void layernorm_rows(
    const float* __restrict__ X, const float* __restrict__ g,
    const float* __restrict__ b, float* __restrict__ Y, __bf16* __restrict__ Ybf)
{
  const int D = H_DIM;
  const int row = blockIdx.x;
  const float* x = X + (size_t)row * D;
  float s = 0.f, s2 = 0.f;
  for (int i = threadIdx.x; i < D; i += 256) { float v = x[i]; s += v; s2 += v * v; }
  __shared__ float sh1[256], sh2[256];
  sh1[threadIdx.x] = s; sh2[threadIdx.x] = s2;
  __syncthreads();
  for (int off = 128; off > 0; off >>= 1) {
    if (threadIdx.x < off) {
      sh1[threadIdx.x] += sh1[threadIdx.x + off];
      sh2[threadIdx.x] += sh2[threadIdx.x + off];
    }
    __syncthreads();
  }
  const float mean = sh1[0] * (1.0f / D);
  const float var  = sh2[0] * (1.0f / D) - mean * mean;
  const float inv  = rsqrtf(var + 1e-5f);
  float*  y  = Y   + (size_t)row * D;
  __bf16* yb = Ybf + (size_t)row * D;
  for (int i = threadIdx.x; i < D; i += 256) {
    float v = (x[i] - mean) * inv * g[i] + b[i];
    y[i] = v;
    yb[i] = f2bf(v);
  }
}

// Self-attention, seq_len=2, 16 heads, head_dim 64; one thread = (batch, head).
// qkv rows: [q(1024) | k(1024) | v(1024)], bf16 in, bf16 out.
__global__ __launch_bounds__(256) void attn_seq2(
    const __bf16* __restrict__ qkv_i, const __bf16* __restrict__ qkv_t,
    __bf16* __restrict__ oi, __bf16* __restrict__ ot)
{
  const int idx = blockIdx.x * blockDim.x + threadIdx.x;
  if (idx >= BATCH * 16) return;
  const int b = idx >> 4, h = idx & 15;
  const __bf16* qi = qkv_i + (size_t)b * 3072 + h * 64;
  const __bf16* ki = qi + 1024;
  const __bf16* vi = qi + 2048;
  const __bf16* qt = qkv_t + (size_t)b * 3072 + h * 64;
  const __bf16* kt = qt + 1024;
  const __bf16* vt = qt + 2048;

  float s00 = 0.f, s01 = 0.f, s10 = 0.f, s11 = 0.f;
  #pragma unroll 8
  for (int d = 0; d < 64; ++d) {
    float a = bf2f(qi[d]), q2 = bf2f(qt[d]), c = bf2f(ki[d]), e = bf2f(kt[d]);
    s00 += a * c; s01 += a * e; s10 += q2 * c; s11 += q2 * e;
  }
  const float sc = 0.125f;             // 1/sqrt(64)
  s00 *= sc; s01 *= sc; s10 *= sc; s11 *= sc;
  const float m0 = fmaxf(s00, s01), m1 = fmaxf(s10, s11);
  const float e00 = expf(s00 - m0), e01 = expf(s01 - m0);
  const float e10 = expf(s10 - m1), e11 = expf(s11 - m1);
  const float z0 = 1.f / (e00 + e01), z1 = 1.f / (e10 + e11);
  const float a00 = e00 * z0, a01 = e01 * z0, a10 = e10 * z1, a11 = e11 * z1;

  __bf16* po = oi + (size_t)b * H_DIM + h * 64;
  __bf16* pt = ot + (size_t)b * H_DIM + h * 64;
  #pragma unroll 8
  for (int d = 0; d < 64; ++d) {
    float x = bf2f(vi[d]), y = bf2f(vt[d]);
    po[d] = f2bf(a00 * x + a01 * y);
    pt[d] = f2bf(a10 * x + a11 * y);
  }
}

// out = 0.5*(fin_i + fin_t) + gate*img_proj + (1-gate)*txt_proj
__global__ __launch_bounds__(256) void final_combine(
    const float* __restrict__ fi, const float* __restrict__ ft,
    const float* __restrict__ gate, const float* __restrict__ ip,
    const float* __restrict__ tp, float* __restrict__ out, int n)
{
  const int i = blockIdx.x * blockDim.x + threadIdx.x;
  if (i < n) {
    const float g = gate[i];
    out[i] = 0.5f * (fi[i] + ft[i]) + g * ip[i] + (1.f - g) * tp[i];
  }
}

template <int ACT, bool HAS_RES, bool WF32, bool WBF>
static inline void gemm(hipStream_t s,
    const __bf16* A, int lda, const __bf16* A2, int kSplit,
    const __bf16* W, const float* bias, const float* Res,
    float* C, __bf16* Cbf, int N, int K)
{
  dim3 grid(N / BN, BATCH / BM);
  gemm_bias_act<ACT, HAS_RES, WF32, WBF>
      <<<grid, 256, 0, s>>>(A, lda, A2, kSplit, W, bias, Res, C, Cbf, N, K);
}

static inline void cvt(hipStream_t s, const float* x, __bf16* y, long n) {
  cvt_bf16<<<(int)((n / 8 + 255) / 256), 256, 0, s>>>(x, y, n);
}

extern "C" void kernel_launch(void* const* d_in, const int* in_sizes, int n_in,
                              void* d_out, int out_size, void* d_ws, size_t ws_size,
                              hipStream_t stream)
{
  (void)in_sizes; (void)n_in; (void)out_size; (void)ws_size;

  const float* img     = (const float*)d_in[0];
  const float* txt     = (const float*)d_in[1];
  const float* Wi      = (const float*)d_in[2];
  const float* bi      = (const float*)d_in[3];
  const float* Wt      = (const float*)d_in[4];
  const float* bt      = (const float*)d_in[5];
  const float* ca_wqkv = (const float*)d_in[6];
  const float* ca_bqkv = (const float*)d_in[7];
  const float* ca_wo   = (const float*)d_in[8];
  const float* ca_bo   = (const float*)d_in[9];
  const float* sa_wqkv = (const float*)d_in[10];
  const float* sa_bqkv = (const float*)d_in[11];
  const float* sa_wo   = (const float*)d_in[12];
  const float* sa_bo   = (const float*)d_in[13];
  const float* n1_g    = (const float*)d_in[14];
  const float* n1_b    = (const float*)d_in[15];
  const float* n2_g    = (const float*)d_in[16];
  const float* n2_b    = (const float*)d_in[17];
  const float* n3_g    = (const float*)d_in[18];
  const float* n3_b    = (const float*)d_in[19];
  const float* ffn_w1  = (const float*)d_in[20];
  const float* ffn_b1  = (const float*)d_in[21];
  const float* ffn_w2  = (const float*)d_in[22];
  const float* ffn_b2  = (const float*)d_in[23];
  const float* gate_w  = (const float*)d_in[24];
  const float* gate_b  = (const float*)d_in[25];

  const int H = H_DIM;
  char* base = (char*)d_ws;
  size_t off = 0;
  auto alloc = [&](size_t bytes) -> void* {
    void* p = base + off;
    off += (bytes + 255) & ~(size_t)255;
    return p;
  };
  const size_t S4 = (size_t)BATCH * H * 4;   // 32 MB f32 slot
  const size_t S2 = (size_t)BATCH * H * 2;   // 16 MB bf16 slot

  // f32 buffers
  float* img_proj = (float*)alloc(S4);
  float* txt_proj = (float*)alloc(S4);
  float* pre_a    = (float*)alloc(S4);
  float* pre_b    = (float*)alloc(S4);
  float* img_att  = (float*)alloc(S4);
  float* txt_att  = (float*)alloc(S4);
  float* ref_i    = (float*)alloc(S4);
  float* ref_t    = (float*)alloc(S4);
  float* gate     = (float*)alloc(S4);
  float* fin_i = img_att;                    // dead after sa-wo residual read
  float* fin_t = txt_att;

  // bf16 inputs/weights
  __bf16* img_bf    = (__bf16*)alloc((size_t)BATCH * 1280 * 2);
  __bf16* txt_bf    = (__bf16*)alloc((size_t)BATCH * 2048 * 2);
  __bf16* Wi_bf     = (__bf16*)alloc((size_t)H * 1280 * 2);
  __bf16* Wt_bf     = (__bf16*)alloc((size_t)H * 2048 * 2);
  __bf16* wv_bf     = (__bf16*)alloc((size_t)H * H * 2);
  __bf16* ca_wo_bf  = (__bf16*)alloc((size_t)H * H * 2);
  __bf16* sa_wqkv_bf= (__bf16*)alloc((size_t)3 * H * H * 2);
  __bf16* sa_wo_bf  = (__bf16*)alloc((size_t)H * H * 2);
  __bf16* ffn_w1_bf = (__bf16*)alloc((size_t)4 * H * H * 2);
  __bf16* ffn_w2_bf = (__bf16*)alloc((size_t)4 * H * H * 2);
  __bf16* gate_w_bf = (__bf16*)alloc((size_t)2 * H * H * 2);

  // bf16 activations
  __bf16* img_proj_bf = (__bf16*)alloc(S2);
  __bf16* txt_proj_bf = (__bf16*)alloc(S2);
  __bf16* tv          = (__bf16*)alloc(S2);   // reused as attn_i
  __bf16* iv          = (__bf16*)alloc(S2);   // reused as attn_t
  __bf16* img_att_bf  = (__bf16*)alloc(S2);
  __bf16* txt_att_bf  = (__bf16*)alloc(S2);
  __bf16* qkv_i_bf    = (__bf16*)alloc(3 * S2);
  __bf16* qkv_t_bf    = (__bf16*)alloc(3 * S2);
  __bf16* ref_i_bf    = (__bf16*)alloc(S2);
  __bf16* ref_t_bf    = (__bf16*)alloc(S2);
  __bf16* h_i         = (__bf16*)alloc(4 * S2);
  __bf16* h_t         = (__bf16*)alloc(4 * S2);
  __bf16* attn_i = tv;
  __bf16* attn_t = iv;

  // --- Pre-convert inputs & weights to bf16 ---------------------------------
  cvt(stream, img, img_bf, (long)BATCH * 1280);
  cvt(stream, txt, txt_bf, (long)BATCH * 2048);
  cvt(stream, Wi, Wi_bf, (long)H * 1280);
  cvt(stream, Wt, Wt_bf, (long)H * 2048);
  cvt(stream, ca_wqkv + (size_t)2 * H * H, wv_bf, (long)H * H);  // only wv used
  cvt(stream, ca_wo, ca_wo_bf, (long)H * H);
  cvt(stream, sa_wqkv, sa_wqkv_bf, (long)3 * H * H);
  cvt(stream, sa_wo, sa_wo_bf, (long)H * H);
  cvt(stream, ffn_w1, ffn_w1_bf, (long)4 * H * H);
  cvt(stream, ffn_w2, ffn_w2_bf, (long)4 * H * H);
  cvt(stream, gate_w, gate_w_bf, (long)2 * H * H);

  const float* ca_bv = ca_bqkv + 2 * H;
  const int BIGK = 1 << 30;

  // --- Input projections (f32 + bf16 outputs) -------------------------------
  gemm<0,false,true,true >(stream, img_bf, 1280, img_bf, BIGK, Wi_bf, bi, nullptr, img_proj, img_proj_bf, H, 1280);
  gemm<0,false,true,true >(stream, txt_bf, 2048, txt_bf, BIGK, Wt_bf, bt, nullptr, txt_proj, txt_proj_bf, H, 2048);

  // --- Cross-attn (seq 1): out = (x@wv^T+bv)@wo^T+bo, + residual, LN --------
  gemm<0,false,false,true>(stream, txt_proj_bf, H, txt_proj_bf, BIGK, wv_bf, ca_bv, nullptr, nullptr, tv, H, H);
  gemm<0,false,false,true>(stream, img_proj_bf, H, img_proj_bf, BIGK, wv_bf, ca_bv, nullptr, nullptr, iv, H, H);
  gemm<0,true ,true ,false>(stream, tv, H, tv, BIGK, ca_wo_bf, ca_bo, img_proj, pre_a, nullptr, H, H);
  gemm<0,true ,true ,false>(stream, iv, H, iv, BIGK, ca_wo_bf, ca_bo, txt_proj, pre_b, nullptr, H, H);
  layernorm_rows<<<BATCH, 256, 0, stream>>>(pre_a, n1_g, n1_b, img_att, img_att_bf);
  layernorm_rows<<<BATCH, 256, 0, stream>>>(pre_b, n2_g, n2_b, txt_att, txt_att_bf);

  // --- Self-attn over [img_att; txt_att] (seq 2) ----------------------------
  gemm<0,false,false,true>(stream, img_att_bf, H, img_att_bf, BIGK, sa_wqkv_bf, sa_bqkv, nullptr, nullptr, qkv_i_bf, 3 * H, H);
  gemm<0,false,false,true>(stream, txt_att_bf, H, txt_att_bf, BIGK, sa_wqkv_bf, sa_bqkv, nullptr, nullptr, qkv_t_bf, 3 * H, H);
  attn_seq2<<<(BATCH * 16) / 256, 256, 0, stream>>>(qkv_i_bf, qkv_t_bf, attn_i, attn_t);
  gemm<0,true ,true ,false>(stream, attn_i, H, attn_i, BIGK, sa_wo_bf, sa_bo, img_att, pre_a, nullptr, H, H);
  gemm<0,true ,true ,false>(stream, attn_t, H, attn_t, BIGK, sa_wo_bf, sa_bo, txt_att, pre_b, nullptr, H, H);
  layernorm_rows<<<BATCH, 256, 0, stream>>>(pre_a, n3_g, n3_b, ref_i, ref_i_bf);
  layernorm_rows<<<BATCH, 256, 0, stream>>>(pre_b, n3_g, n3_b, ref_t, ref_t_bf);

  // --- FFN with residual ----------------------------------------------------
  gemm<1,false,false,true>(stream, ref_i_bf, H, ref_i_bf, BIGK, ffn_w1_bf, ffn_b1, nullptr, nullptr, h_i, 4 * H, H);
  gemm<1,false,false,true>(stream, ref_t_bf, H, ref_t_bf, BIGK, ffn_w1_bf, ffn_b1, nullptr, nullptr, h_t, 4 * H, H);
  gemm<0,true ,true ,false>(stream, h_i, 4 * H, h_i, BIGK, ffn_w2_bf, ffn_b2, ref_i, fin_i, nullptr, H, 4 * H);
  gemm<0,true ,true ,false>(stream, h_t, 4 * H, h_t, BIGK, ffn_w2_bf, ffn_b2, ref_t, fin_t, nullptr, H, 4 * H);

  // --- gate = sigmoid([img_proj | txt_proj] @ gate_w^T + gate_b) ------------
  gemm<2,false,true ,false>(stream, img_proj_bf, H, txt_proj_bf, H, gate_w_bf, gate_b, nullptr, gate, nullptr, H, 2 * H);

  // --- out = mean(refined) + gate*img_proj + (1-gate)*txt_proj --------------
  final_combine<<<(BATCH * H) / 256, 256, 0, stream>>>(
      fin_i, fin_t, gate, img_proj, txt_proj, (float*)d_out, BATCH * H);
}
